// SelfAttention_25898652795192
// MI455X (gfx1250) — matrix operand order repaired and checked by reading, not compiled, not run on hardware
//
#include <hip/hip_runtime.h>

#define HEADS 16
#define HDIM  64
#define EMB   1024
#define BATCH 4
#define SEQ   2048

typedef __attribute__((ext_vector_type(16))) _Float16 v16h;
typedef __attribute__((ext_vector_type(8)))  float    v8f;

#define DI __device__ __forceinline__

// D = A(16x32 f16) * B(32x16 f16) + C(16x16 f32), wave32 WMMA
DI v8f wmma16(v16h a, v16h b, v8f c) {
  return __builtin_amdgcn_wmma_f32_16x16x32_f16(false, a, false, b, (short)0, c,
                                                false, false);
}

// A fragment (16x32, 16-bit): lane 0-15 -> M=lane, lanes 16-31 -> M=lane-16.
// VGPR v, half h: K = (v>>2)*16 + (lane>=16)*8 + (v&3)*2 + h  (ISA 7.12.2 table)
DI int a_koff(int p, int hi) { return ((p >> 2) << 4) + (hi << 3) + ((p & 3) << 1); }
// B fragment (32x16, 16-bit): lane holds column N=lane&15; same K order as A:
// K = (v>>2)*16 + (lane>=16)*8 + (v&3)*2 + h
DI int b_koff(int p, int hi) { return ((p >> 2) << 4) + (hi << 3) + ((p & 3) << 1); }

// Load A fragment from f16, element (m,k) at base[m*ld + k]; contiguous K pairs
DI v16h load_a_h(const _Float16* base, int ld) {
  int l = threadIdx.x & 31, hi = l >> 4, m = l & 15;
  union { v16h v; unsigned u[8]; } r;
  const _Float16* row = base + (long)m * ld;
#pragma unroll
  for (int p = 0; p < 8; ++p)
    r.u[p] = *(const unsigned*)(row + a_koff(p, hi));
  return r.v;
}
// Load B fragment from f16, element (k,n) at base[n*ld + k]
DI v16h load_b_h(const _Float16* base, int ld) {
  int l = threadIdx.x & 31, hi = l >> 4, n = l & 15;
  union { v16h v; unsigned u[8]; } r;
  const _Float16* col = base + (long)n * ld;
#pragma unroll
  for (int p = 0; p < 8; ++p)
    r.u[p] = *(const unsigned*)(col + b_koff(p, hi));
  return r.v;
}
// Same, but source is fp32 (convert on load)
DI v16h load_a_f(const float* base, int ld) {
  int l = threadIdx.x & 31, hi = l >> 4, m = l & 15;
  v16h r;
  const float* row = base + (long)m * ld;
#pragma unroll
  for (int p = 0; p < 8; ++p) {
    int k = a_koff(p, hi);
    r[2 * p]     = (_Float16)row[k];
    r[2 * p + 1] = (_Float16)row[k + 1];
  }
  return r;
}
DI v16h load_b_f(const float* base, int ld) {
  int l = threadIdx.x & 31, hi = l >> 4, n = l & 15;
  v16h r;
  const float* col = base + (long)n * ld;
#pragma unroll
  for (int p = 0; p < 8; ++p) {
    int k = b_koff(p, hi);
    r[2 * p]     = (_Float16)col[k];
    r[2 * p + 1] = (_Float16)col[k + 1];
  }
  return r;
}

// ---------------- Wfc fp32 -> f16 ----------------
__global__ void cvt_kernel(const float* __restrict__ W, _Float16* __restrict__ Wh,
                           int n) {
  int i = blockIdx.x * blockDim.x + threadIdx.x;
  if (i < n) Wh[i] = (_Float16)W[i];
}

// ---------------- Per-head projection: rows of [B*L*H, 64] @ W^T + b -> f16 ----
// out layout: transposeV==0 -> [b,h,l,64] ; transposeV==1 -> [b,h,64,l]
__global__ __launch_bounds__(128) void proj_kernel(
    const float* __restrict__ X, const float* __restrict__ W,
    const float* __restrict__ bias, _Float16* __restrict__ out, int transposeV) {
  int wave = threadIdx.x >> 5;
  long R0 = ((long)blockIdx.x * 4 + wave) * 16;  // 16-row tile of (b,l,h) rows
  int l = threadIdx.x & 31, hi = l >> 4, n = l & 15;

  v16h a0 = load_a_f(X + R0 * HDIM, HDIM);        // d 0..31
  v16h a1 = load_a_f(X + R0 * HDIM + 32, HDIM);   // d 32..63

#pragma unroll
  for (int et = 0; et < 4; ++et) {                // 4 output column tiles of 16
    v8f c = {};
    c = wmma16(a0, load_b_f(W + et * 16 * HDIM, HDIM), c);       // W[e][d], d 0..31
    c = wmma16(a1, load_b_f(W + et * 16 * HDIM + 32, HDIM), c);  // d 32..63
    float be = bias[et * 16 + n];
#pragma unroll
    for (int v = 0; v < 8; ++v) {
      long R = R0 + v + 8 * hi;
      int h   = (int)(R & (HEADS - 1));
      int seq = (int)((R >> 4) & (SEQ - 1));
      int b   = (int)(R >> 15);                   // L*H = 2^15 rows per batch
      int e   = et * 16 + n;
      long bh = (long)b * HEADS + h;
      long idx = transposeV ? (bh * HDIM + e) * SEQ + seq
                            : (bh * SEQ + seq) * HDIM + e;
      out[idx] = (_Float16)(c[v] + be);
    }
  }
}

// ---------------- Flash attention (one wave = 16 query rows of one (b,h)) ----
__global__ __launch_bounds__(128) void attn_kernel(
    const _Float16* __restrict__ Qp, const _Float16* __restrict__ Kp,
    const _Float16* __restrict__ Vt, const int* __restrict__ mask,
    _Float16* __restrict__ Oa) {
  __shared__ _Float16 pbuf[4][16][36];            // per-wave P transpose buffer
  int wave = threadIdx.x >> 5;
  int wid  = blockIdx.x * 4 + wave;
  int qt   = wid & 127;                           // 128 q-tiles per (b,h)
  int bh   = wid >> 7;
  int b    = bh >> 4;
  int h    = bh & (HEADS - 1);
  int l = threadIdx.x & 31, hi = l >> 4, n = l & 15;

  const _Float16* qbase = Qp + ((long)bh * SEQ + qt * 16) * HDIM;
  v16h aq0 = load_a_h(qbase, HDIM);
  v16h aq1 = load_a_h(qbase + 32, HDIM);

  v8f o0 = {}, o1 = {}, o2 = {}, o3 = {};
  float rm[8], rl[8];
#pragma unroll
  for (int v = 0; v < 8; ++v) { rm[v] = -__builtin_inff(); rl[v] = 0.f; }

  const int* mrow = mask + (long)b * SEQ;
  const _Float16* kroot = Kp + (long)bh * SEQ * HDIM;
  const _Float16* vroot = Vt + (long)bh * HDIM * SEQ;

  for (int kb = 0; kb < SEQ; kb += 32) {
    const _Float16* kbase = kroot + (long)kb * HDIM;
    if (kb + 32 < SEQ) {                          // gfx1250 global_prefetch_b8
      __builtin_prefetch(kbase + 32 * HDIM, 0, 0);
      __builtin_prefetch(vroot + kb + 32, 0, 0);
    }
    // S = Q K^T for two 16-key subtiles (K stored [key][d])
    v8f s0 = {}, s1 = {};
    s0 = wmma16(aq0, load_b_h(kbase, HDIM), s0);
    s0 = wmma16(aq1, load_b_h(kbase + 32, HDIM), s0);
    s1 = wmma16(aq0, load_b_h(kbase + 16 * HDIM, HDIM), s1);
    s1 = wmma16(aq1, load_b_h(kbase + 16 * HDIM + 32, HDIM), s1);

    int mk0 = mrow[kb + n];
    int mk1 = mrow[kb + 16 + n];
    float p0[8], p1[8];
#pragma unroll
    for (int v = 0; v < 8; ++v) {
      float x0 = mk0 ? s0[v] : -1e20f;            // mask, then floor(s/8)
      float x1 = mk1 ? s1[v] : -1e20f;
      p0[v] = floorf(x0 * 0.125f);
      p1[v] = floorf(x1 * 0.125f);
    }
#pragma unroll
    for (int v = 0; v < 8; ++v) {
      float bm = fmaxf(p0[v], p1[v]);             // row max over 16 lanes
      bm = fmaxf(bm, __shfl_xor(bm, 1, 32));
      bm = fmaxf(bm, __shfl_xor(bm, 2, 32));
      bm = fmaxf(bm, __shfl_xor(bm, 4, 32));
      bm = fmaxf(bm, __shfl_xor(bm, 8, 32));
      float mn = fmaxf(rm[v], bm);
      float sc = __expf(rm[v] - mn);
      rm[v] = mn;
      float e0 = __expf(p0[v] - mn);
      float e1 = __expf(p1[v] - mn);
      float ps = e0 + e1;
      ps += __shfl_xor(ps, 1, 32);
      ps += __shfl_xor(ps, 2, 32);
      ps += __shfl_xor(ps, 4, 32);
      ps += __shfl_xor(ps, 8, 32);
      rl[v] = rl[v] * sc + ps;
      o0[v] *= sc; o1[v] *= sc; o2[v] *= sc; o3[v] *= sc;
      int mr = v + 8 * hi;                        // C layout -> row-major LDS
      pbuf[wave][mr][n]      = (_Float16)e0;
      pbuf[wave][mr][16 + n] = (_Float16)e1;
    }
    // same-wave LDS RAW across lanes: DS ops are in-order per wave
    asm volatile("s_wait_dscnt 0" ::: "memory");

    v16h ap = load_a_h(&pbuf[wave][0][0], 36);    // P as A fragment (16x32)
    const _Float16* vbase = vroot + kb;           // Vt stored [d][seq]
    o0 = wmma16(ap, load_b_h(vbase + 0 * 16 * SEQ, SEQ), o0);
    o1 = wmma16(ap, load_b_h(vbase + 1 * 16 * SEQ, SEQ), o1);
    o2 = wmma16(ap, load_b_h(vbase + 2 * 16 * SEQ, SEQ), o2);
    o3 = wmma16(ap, load_b_h(vbase + 3 * 16 * SEQ, SEQ), o3);
  }

#pragma unroll
  for (int v = 0; v < 8; ++v) {
    float iv = 1.0f / rl[v];
    long row = (long)b * SEQ + qt * 16 + v + 8 * hi;
    long base = row * EMB + h * HDIM + n;
    Oa[base + 0]  = (_Float16)(o0[v] * iv);
    Oa[base + 16] = (_Float16)(o1[v] * iv);
    Oa[base + 32] = (_Float16)(o2[v] * iv);
    Oa[base + 48] = (_Float16)(o3[v] * iv);
  }
}

// ---------------- Final FC: [B*L,1024] @ Wfc^T + bfc -> fp32 -----------------
__global__ __launch_bounds__(128) void fc_kernel(
    const _Float16* __restrict__ Oa, const _Float16* __restrict__ W16,
    const float* __restrict__ bfc, float* __restrict__ out) {
  int wave = threadIdx.x >> 5;
  int wid  = blockIdx.x * 4 + wave;
  int nt   = wid & 15;                            // 16 column tiles of 64
  int mt   = wid >> 4;                            // 512 row tiles of 16
  int l = threadIdx.x & 31, hi = l >> 4, n = l & 15;

  v8f c0 = {}, c1 = {}, c2 = {}, c3 = {};
  const _Float16* arow = Oa  + (long)mt * 16 * EMB;
  const _Float16* wrow = W16 + (long)nt * 64 * EMB;  // Wfc[e][d] row-major
  for (int kb = 0; kb < EMB; kb += 32) {
    v16h a = load_a_h(arow + kb, EMB);
    c0 = wmma16(a, load_b_h(wrow + kb, EMB), c0);
    c1 = wmma16(a, load_b_h(wrow + 16 * EMB + kb, EMB), c1);
    c2 = wmma16(a, load_b_h(wrow + 32 * EMB + kb, EMB), c2);
    c3 = wmma16(a, load_b_h(wrow + 48 * EMB + kb, EMB), c3);
  }
  float b0 = bfc[nt * 64 + n],      b1 = bfc[nt * 64 + 16 + n];
  float b2 = bfc[nt * 64 + 32 + n], b3 = bfc[nt * 64 + 48 + n];
#pragma unroll
  for (int v = 0; v < 8; ++v) {
    long r = (long)(mt * 16 + v + 8 * hi) * EMB + nt * 64 + n;
    out[r + 0]  = c0[v] + b0;
    out[r + 16] = c1[v] + b1;
    out[r + 32] = c2[v] + b2;
    out[r + 48] = c3[v] + b3;
  }
}

extern "C" void kernel_launch(void* const* d_in, const int* in_sizes, int n_in,
                              void* d_out, int out_size, void* d_ws, size_t ws_size,
                              hipStream_t stream) {
  const float* q    = (const float*)d_in[0];
  const float* k    = (const float*)d_in[1];
  const float* v    = (const float*)d_in[2];
  const int*   mask = (const int*)d_in[3];
  const float* Wq   = (const float*)d_in[4];
  const float* bq   = (const float*)d_in[5];
  const float* Wk   = (const float*)d_in[6];
  const float* bk   = (const float*)d_in[7];
  const float* Wv   = (const float*)d_in[8];
  const float* bv   = (const float*)d_in[9];
  const float* Wfc  = (const float*)d_in[10];
  const float* bfc  = (const float*)d_in[11];
  float* out = (float*)d_out;

  const long SZ = (long)BATCH * HEADS * SEQ * HDIM;  // 8,388,608 f16 elements
  _Float16* Qp  = (_Float16*)d_ws;       // [b,h,l,64]
  _Float16* Kp  = Qp + SZ;               // [b,h,l,64]
  _Float16* Vt  = Kp + SZ;               // [b,h,64,l]
  _Float16* Oa  = Vt + SZ;               // [b*l, 1024]
  _Float16* W16 = Oa + SZ;               // Wfc in f16, [1024,1024]

  cvt_kernel<<<(EMB * EMB) / 256, 256, 0, stream>>>(Wfc, W16, EMB * EMB);

  int projBlocks = (BATCH * SEQ * HEADS) / 16 / 4;   // 2048
  proj_kernel<<<projBlocks, 128, 0, stream>>>(q, Wq, bq, Qp, 0);
  proj_kernel<<<projBlocks, 128, 0, stream>>>(k, Wk, bk, Kp, 0);
  proj_kernel<<<projBlocks, 128, 0, stream>>>(v, Wv, bv, Vt, 1);

  attn_kernel<<<(BATCH * HEADS * (SEQ / 16)) / 4, 128, 0, stream>>>(Qp, Kp, Vt,
                                                                    mask, Oa);
  fc_kernel<<<((BATCH * SEQ / 16) * (EMB / 64)) / 4, 128, 0, stream>>>(Oa, W16,
                                                                       bfc, out);
}
